// GGNN_88252987998920
// MI455X (gfx1250) — matrix operand order repaired and checked by reading, not compiled, not run on hardware
//
#include <hip/hip_runtime.h>
#include <math.h>

// ---------------- problem constants ----------------
constexpr int kB = 4;
constexpr int kN = 10000;
constexpr int kD = 256;
constexpr int kS = 2;
constexpr int kE = 160000;
constexpr int kPASSES = 6;
constexpr int kHID = 80;
constexpr int kOUT = 10;
constexpr int kNK = kD / 32;       // K-steps of 32 per 256-deep GEMM

// ---------------- WMMA types ----------------
typedef __attribute__((ext_vector_type(16))) __bf16          v16bf;
typedef __attribute__((ext_vector_type(8)))  float           v8f;
typedef __attribute__((ext_vector_type(16))) unsigned short  v16u;

__device__ __forceinline__ v8f wmma_bf16(v16bf a, v16bf b, v8f c) {
  // D = A(16x32) * B(32x16) + C, f32 accumulate
  return __builtin_amdgcn_wmma_f32_16x16x32_bf16(
      false, a, false, b, (short)0, c, false, false);
}

__device__ __forceinline__ unsigned short f2bf(float x) {
  unsigned int u = __builtin_bit_cast(unsigned int, x);
  unsigned int r = u + 0x7FFFu + ((u >> 16) & 1u);   // round-to-nearest-even
  return (unsigned short)(r >> 16);
}

// A fragment (16x32 bf16, ISA layout) from a bf16 row in LDS/global.
// 'row' points at element [m][k0]; kb = 0 (lanes 0-15) or 8 (lanes 16-31).
__device__ __forceinline__ v16bf load_a_bf16(const unsigned short* row, int kb) {
  union { v16u u; uint4 q[2]; } t;
  t.q[0] = *reinterpret_cast<const uint4*>(row + kb);        // K kb..kb+7
  t.q[1] = *reinterpret_cast<const uint4*>(row + kb + 16);   // K kb+16..kb+23
  return __builtin_bit_cast(v16bf, t.u);
}

// A fragment built on the fly from an f32 row in global memory.
__device__ __forceinline__ v16bf load_a_f32(const float* p) {
  union { v16u u; unsigned short s[16]; } t;
  float4 q0 = reinterpret_cast<const float4*>(p)[0];
  float4 q1 = reinterpret_cast<const float4*>(p)[1];
  float4 q2 = reinterpret_cast<const float4*>(p + 16)[0];
  float4 q3 = reinterpret_cast<const float4*>(p + 16)[1];
  t.s[0]=f2bf(q0.x); t.s[1]=f2bf(q0.y); t.s[2]=f2bf(q0.z); t.s[3]=f2bf(q0.w);
  t.s[4]=f2bf(q1.x); t.s[5]=f2bf(q1.y); t.s[6]=f2bf(q1.z); t.s[7]=f2bf(q1.w);
  t.s[8]=f2bf(q2.x); t.s[9]=f2bf(q2.y); t.s[10]=f2bf(q2.z); t.s[11]=f2bf(q2.w);
  t.s[12]=f2bf(q3.x); t.s[13]=f2bf(q3.y); t.s[14]=f2bf(q3.z); t.s[15]=f2bf(q3.w);
  return __builtin_bit_cast(v16bf, t.u);
}

// B fragment: 16 contiguous bf16 per lane (fragment-major pre-packed weights).
__device__ __forceinline__ v16bf load_b_frag(const unsigned short* p) {
  union { v16u u; uint4 q[2]; } t;
  t.q[0] = reinterpret_cast<const uint4*>(p)[0];
  t.q[1] = reinterpret_cast<const uint4*>(p)[1];
  return __builtin_bit_cast(v16bf, t.u);
}

// ---------------- weight pre-pack ----------------
// W is (F x 256) row-major f32, out is fragment-major bf16:
// out[(((nt*kNK)+k0i)*32 + lane)*16 + i] = bf16( W[nt*16 + lane%16][k0i*32 + (lane/16)*8 + i + (i/8)*8] )
__global__ __launch_bounds__(256) void pack_kernel(
    const float* __restrict__ W, unsigned short* __restrict__ out, int F) {
  int t = blockIdx.x * blockDim.x + threadIdx.x;
  int total = (F / 16) * kNK * 32;
  if (t >= total) return;
  int lane = t & 31;
  int k0i  = (t >> 5) & (kNK - 1);
  int nt   = t >> 8;
  int n      = nt * 16 + (lane & 15);
  int kstart = k0i * 32 + ((lane >> 4) << 3);
  union { unsigned short s[16]; uint4 q[2]; } tmp;
#pragma unroll
  for (int i = 0; i < 16; ++i)
    tmp.s[i] = f2bf(W[(size_t)n * kD + kstart + i + ((i >> 3) << 3)]);
  uint4* o = reinterpret_cast<uint4*>(out + (size_t)t * 16);
  o[0] = tmp.q[0];
  o[1] = tmp.q[1];
}

// ---------------- edge MLP + scatter ----------------
// One block: 16 edges (full 256 features). 8 waves, 2 N-tiles per wave.
__global__ __launch_bounds__(256) void edge_mlp_kernel(
    const float* __restrict__ h, const int* __restrict__ edges,
    const unsigned short* __restrict__ pw1, const float* __restrict__ b1,
    const unsigned short* __restrict__ pw2, const float* __restrict__ b2,
    float* __restrict__ incoming, int si) {
  __shared__ unsigned short Xs[16 * kD];   // gathered rows (bf16)
  __shared__ unsigned short Ys[16 * kD];   // layer-1 activations (bf16)
  __shared__ int s_dst[16];
  __shared__ int s_src[16];

  const int b   = blockIdx.y;
  const int e0  = blockIdx.x * 16;
  const int tid = threadIdx.x;
  const int lane = tid & 31;
  const int wave = tid >> 5;

  if (tid < 16) {
    const int* ep = edges + (((size_t)b * kS + si) * kE + (e0 + tid)) * 2;
    s_dst[tid] = ep[0];
    s_src[tid] = ep[1];
  }
  __syncthreads();

  // gather + f32->bf16 into LDS: 256 threads * 16 floats
  {
    int row = tid >> 4;
    int cc  = (tid & 15) * 16;
    const float* srcp = h + ((size_t)b * kN + s_src[row]) * kD + cc;
    union { unsigned short s[16]; uint4 q[2]; } tmp;
#pragma unroll
    for (int i = 0; i < 16; i += 4) {
      float4 v = *reinterpret_cast<const float4*>(srcp + i);
      tmp.s[i+0] = f2bf(v.x); tmp.s[i+1] = f2bf(v.y);
      tmp.s[i+2] = f2bf(v.z); tmp.s[i+3] = f2bf(v.w);
    }
    uint4* d = reinterpret_cast<uint4*>(&Xs[row * kD + cc]);
    d[0] = tmp.q[0];
    d[1] = tmp.q[1];
  }
  __syncthreads();

  const int m    = lane & 15;            // A row / C column id
  const int kb   = (lane < 16) ? 0 : 8;  // A K-base
  const int ncol = lane & 15;
  const int nt0  = wave * 2;
  const int nt1  = nt0 + 1;
  const int mhi  = (lane >> 4) << 3;     // C row offset (0 or 8)

  // ---- GEMM 1: Y = leaky(X * W1^T + b1) ----
  {
    v8f c0 = {}; v8f c1 = {};
    const unsigned short* wb = pw1 + (size_t)si * kD * kD;
#pragma unroll
    for (int k = 0; k < kNK; ++k) {
      v16bf a  = load_a_bf16(&Xs[m * kD + k * 32], kb);
      v16bf b0 = load_b_frag(wb + (((size_t)nt0 * kNK + k) * 32 + lane) * 16);
      v16bf b1f = load_b_frag(wb + (((size_t)nt1 * kNK + k) * 32 + lane) * 16);
      c0 = wmma_bf16(a, b0, c0);
      c1 = wmma_bf16(a, b1f, c1);
    }
    const float bias0 = b1[si * kD + nt0 * 16 + ncol];
    const float bias1 = b1[si * kD + nt1 * 16 + ncol];
#pragma unroll
    for (int r = 0; r < 8; ++r) {
      int mm = r + mhi;
      float v0 = c0[r] + bias0; v0 = v0 > 0.f ? v0 : 0.01f * v0;
      float v1 = c1[r] + bias1; v1 = v1 > 0.f ? v1 : 0.01f * v1;
      Ys[mm * kD + nt0 * 16 + ncol] = f2bf(v0);
      Ys[mm * kD + nt1 * 16 + ncol] = f2bf(v1);
    }
  }
  __syncthreads();

  // ---- GEMM 2: Z = Y * W2^T + b2, scatter-add to incoming[dst] ----
  {
    v8f c0 = {}; v8f c1 = {};
    const unsigned short* wb = pw2 + (size_t)si * kD * kD;
#pragma unroll
    for (int k = 0; k < kNK; ++k) {
      v16bf a  = load_a_bf16(&Ys[m * kD + k * 32], kb);
      v16bf b0 = load_b_frag(wb + (((size_t)nt0 * kNK + k) * 32 + lane) * 16);
      v16bf b1f = load_b_frag(wb + (((size_t)nt1 * kNK + k) * 32 + lane) * 16);
      c0 = wmma_bf16(a, b0, c0);
      c1 = wmma_bf16(a, b1f, c1);
    }
    const float bias0 = b2[si * kD + nt0 * 16 + ncol];
    const float bias1 = b2[si * kD + nt1 * 16 + ncol];
#pragma unroll
    for (int r = 0; r < 8; ++r) {
      int mm = r + mhi;
      size_t base = ((size_t)b * kN + s_dst[mm]) * kD;
      atomicAdd(&incoming[base + nt0 * 16 + ncol], c0[r] + bias0);
      atomicAdd(&incoming[base + nt1 * 16 + ncol], c1[r] + bias1);
    }
  }
}

// ---------------- fused GRU cell ----------------
// One wave per (16 rows x 16 output features); computes all 6 gate tiles.
__global__ __launch_bounds__(32) void gru_kernel(
    const float* __restrict__ x,   // incoming
    const float* __restrict__ hin,
    const unsigned short* __restrict__ pwih,
    const unsigned short* __restrict__ pwhh,
    const float* __restrict__ bih, const float* __restrict__ bhh,
    float* __restrict__ hout) {
  const int rowt = blockIdx.x;          // 0..624
  const int nt   = blockIdx.y;          // 0..15 feature tile
  const int b    = blockIdx.z;
  const int lane = threadIdx.x;
  const int m    = lane & 15;
  const int kb   = (lane < 16) ? 0 : 8;
  const int ncol = lane & 15;
  const int mhi  = (lane >> 4) << 3;

  const size_t rowbase = ((size_t)b * kN + (size_t)rowt * 16 + m) * kD;

  v8f cir = {}, ciz = {}, cin = {}, chr_ = {}, chz = {}, chn = {};
#pragma unroll
  for (int k = 0; k < kNK; ++k) {
    v16bf ax = load_a_f32(x   + rowbase + k * 32 + kb);
    v16bf ah = load_a_f32(hin + rowbase + k * 32 + kb);
    v16bf br = load_b_frag(pwih + (((size_t)(nt     ) * kNK + k) * 32 + lane) * 16);
    v16bf bz = load_b_frag(pwih + (((size_t)(nt + 16) * kNK + k) * 32 + lane) * 16);
    v16bf bn = load_b_frag(pwih + (((size_t)(nt + 32) * kNK + k) * 32 + lane) * 16);
    v16bf hr = load_b_frag(pwhh + (((size_t)(nt     ) * kNK + k) * 32 + lane) * 16);
    v16bf hz = load_b_frag(pwhh + (((size_t)(nt + 16) * kNK + k) * 32 + lane) * 16);
    v16bf hn = load_b_frag(pwhh + (((size_t)(nt + 32) * kNK + k) * 32 + lane) * 16);
    cir  = wmma_bf16(ax, br, cir);
    ciz  = wmma_bf16(ax, bz, ciz);
    cin  = wmma_bf16(ax, bn, cin);
    chr_ = wmma_bf16(ah, hr, chr_);
    chz  = wmma_bf16(ah, hz, chz);
    chn  = wmma_bf16(ah, hn, chn);
  }

  const int f = nt * 16 + ncol;
  const float bir = bih[f], biz = bih[kD + f], bin = bih[2 * kD + f];
  const float bhr = bhh[f], bhz = bhh[kD + f], bhn = bhh[2 * kD + f];
#pragma unroll
  for (int r = 0; r < 8; ++r) {
    int mm = r + mhi;
    size_t idx = ((size_t)b * kN + (size_t)rowt * 16 + mm) * kD + f;
    float hold = hin[idx];
    float rg = 1.f / (1.f + __expf(-(cir[r] + bir + chr_[r] + bhr)));
    float zg = 1.f / (1.f + __expf(-(ciz[r] + biz + chz[r] + bhz)));
    float ng = tanhf(cin[r] + bin + rg * (chn[r] + bhn));
    hout[idx] = (1.f - zg) * ng + zg * hold;
  }
}

// ---------------- node-sum reduction ----------------
__global__ __launch_bounds__(256) void reduce_kernel(
    const float* __restrict__ h, float* __restrict__ s) {
  const int b = blockIdx.x;
  const int d = threadIdx.x;            // 256 features
  const int n0 = blockIdx.y * 200;      // 50 blocks * 200 rows = 10000
  float acc = 0.f;
  for (int n = n0; n < n0 + 200; ++n)
    acc += h[((size_t)b * kN + n) * kD + d];
  atomicAdd(&s[b * kD + d], acc);
}

// ---------------- readout head ----------------
__global__ __launch_bounds__(256) void head_kernel(
    const float* __restrict__ s, const float* __restrict__ ptype,
    const float* __restrict__ fc1w, const float* __restrict__ fc1b,
    const float* __restrict__ fc2w, const float* __restrict__ fc2b,
    const float* __restrict__ fc3w, const float* __restrict__ fc3b,
    float* __restrict__ out) {
  __shared__ float l[kB][kD + 1];
  __shared__ float x1[kB][kHID];
  __shared__ float x2[kB][kHID];
  __shared__ float fmax[kB];
  const int t = threadIdx.x;

  for (int idx = t; idx < kB * kD; idx += 256) {
    int b = idx >> 8, d = idx & 255;
    float v = logf(s[b * kD + d]);
    if (v != v) v = 0.f;          // NaN -> 0
    v = v > 0.f ? v : 0.f;        // relu (handles -inf)
    l[b][d] = v;
  }
  __syncthreads();
  if (t < kB) {
    float mx = -__builtin_inff();
    for (int d = 0; d < kD; ++d) {
      float v = l[t][d];
      if (v != __builtin_inff() && v > mx) mx = v;
    }
    fmax[t] = mx;
    l[t][kD] = ptype[t];
  }
  __syncthreads();
  for (int idx = t; idx < kB * kD; idx += 256) {
    int b = idx >> 8, d = idx & 255;
    if (l[b][d] == __builtin_inff()) l[b][d] = fmax[b];
  }
  __syncthreads();
  for (int idx = t; idx < kB * kHID; idx += 256) {
    int b = idx / kHID, o = idx % kHID;
    float acc = fc1b[o];
    for (int d = 0; d < kD + 1; ++d) acc += fc1w[o * (kD + 1) + d] * l[b][d];
    x1[b][o] = acc > 0.f ? acc : 0.01f * acc;
  }
  __syncthreads();
  for (int idx = t; idx < kB * kHID; idx += 256) {
    int b = idx / kHID, o = idx % kHID;
    float acc = fc2b[o];
    for (int d = 0; d < kHID; ++d) acc += fc2w[o * kHID + d] * x1[b][d];
    x2[b][o] = acc > 0.f ? acc : 0.01f * acc;
  }
  __syncthreads();
  for (int idx = t; idx < kB * kOUT; idx += 256) {
    int b = idx / kOUT, o = idx % kOUT;
    float acc = fc3b[o];
    for (int d = 0; d < kHID; ++d) acc += fc3w[o * kHID + d] * x2[b][d];
    out[b * kOUT + o] = acc;
  }
}

// ---------------- host launcher ----------------
extern "C" void kernel_launch(void* const* d_in, const int* in_sizes, int n_in,
                              void* d_out, int out_size, void* d_ws, size_t ws_size,
                              hipStream_t stream) {
  const float* nodes = (const float*)d_in[0];
  const int*   edges = (const int*)d_in[1];
  const float* ptype = (const float*)d_in[2];
  const float* gwih  = (const float*)d_in[3];
  const float* gwhh  = (const float*)d_in[4];
  const float* gbih  = (const float*)d_in[5];
  const float* gbhh  = (const float*)d_in[6];
  const float* ew1   = (const float*)d_in[7];
  const float* eb1   = (const float*)d_in[8];
  const float* ew2   = (const float*)d_in[9];
  const float* eb2   = (const float*)d_in[10];
  const float* fc1w  = (const float*)d_in[11];
  const float* fc1b  = (const float*)d_in[12];
  const float* fc2w  = (const float*)d_in[13];
  const float* fc2b  = (const float*)d_in[14];
  const float* fc3w  = (const float*)d_in[15];
  const float* fc3b  = (const float*)d_in[16];
  float* out = (float*)d_out;

  char* ws = (char*)d_ws;
  const size_t HB = (size_t)kB * kN * kD * sizeof(float);    // 40.96 MB
  float* h0   = (float*)(ws);
  float* h1   = (float*)(ws + HB);
  float* inc  = (float*)(ws + 2 * HB);
  float* sbuf = (float*)(ws + 3 * HB);
  unsigned short* pw1 = (unsigned short*)(ws + 3 * HB + 4096);
  unsigned short* pw2 = pw1 + (size_t)kS * kD * kD;
  unsigned short* pih = pw2 + (size_t)kS * kD * kD;
  unsigned short* phh = pih + (size_t)3 * kD * kD;

  // pack all weights to fragment-major bf16 (cheap, deterministic)
  {
    int tE = kD * 16;       // threads per 256x256 matrix
    pack_kernel<<<(tE + 255) / 256, 256, 0, stream>>>(ew1,            pw1,            kD);
    pack_kernel<<<(tE + 255) / 256, 256, 0, stream>>>(ew1 + kD * kD,  pw1 + kD * kD,  kD);
    pack_kernel<<<(tE + 255) / 256, 256, 0, stream>>>(ew2,            pw2,            kD);
    pack_kernel<<<(tE + 255) / 256, 256, 0, stream>>>(ew2 + kD * kD,  pw2 + kD * kD,  kD);
    int tG = 3 * kD * 16;   // threads per 768x256 matrix
    pack_kernel<<<(tG + 255) / 256, 256, 0, stream>>>(gwih, pih, 3 * kD);
    pack_kernel<<<(tG + 255) / 256, 256, 0, stream>>>(gwhh, phh, 3 * kD);
  }

  hipMemcpyAsync(h0, nodes, HB, hipMemcpyDeviceToDevice, stream);

  float* hc = h0;
  float* hn = h1;
  for (int j = 0; j < kPASSES; ++j) {
    hipMemsetAsync(inc, 0, HB, stream);
    dim3 eg(kE / 16, kB);
    edge_mlp_kernel<<<eg, 256, 0, stream>>>(hc, edges, pw1, eb1, pw2, eb2, inc, 0);
    edge_mlp_kernel<<<eg, 256, 0, stream>>>(hc, edges, pw1, eb1, pw2, eb2, inc, 1);
    dim3 gg(kN / 16, 16, kB);
    gru_kernel<<<gg, 32, 0, stream>>>(inc, hc, pih, phh, gbih, gbhh, hn);
    float* tp = hc; hc = hn; hn = tp;
  }

  hipMemsetAsync(sbuf, 0, kB * kD * sizeof(float), stream);
  dim3 rg(kB, 50);
  reduce_kernel<<<rg, 256, 0, stream>>>(hc, sbuf);
  head_kernel<<<1, 256, 0, stream>>>(sbuf, ptype, fc1w, fc1b, fc2w, fc2b,
                                     fc3w, fc3b, out);
}
